// NodeEdge_9259949490935
// MI455X (gfx1250) — matrix operand order repaired and checked by reading, not compiled, not run on hardware
//
#include <hip/hip_runtime.h>

// Problem constants (reference: N=4096, E=8192, F=256, B=16)
#define N_ 4096
#define E_ 8192
#define F_ 256
#define B_ 16

typedef __attribute__((ext_vector_type(16))) __bf16 v16bf;
typedef __attribute__((ext_vector_type(8)))  float  v8f;

// Tiling: 128x128 output tile per block, K-chunk 64, 8 waves (wave32).
constexpr int TN = 128;
constexpr int TC = 128;
constexpr int TK = 64;
constexpr int KP = 72;   // LDS row pitch in bf16: 144 B = 16B aligned, 36-dword stride (odd/64 spread)

union Frag32 {           // one WMMA bf16 operand: 16 bf16 per lane = 32 bytes
    v16bf v;
    uint4 q[2];
};

__device__ __forceinline__ unsigned short f32_to_bf16(float f) {
    unsigned int u = __float_as_uint(f);
    // round-to-nearest-even truncation to bf16
    u += 0x7FFFu + ((u >> 16) & 1u);
    return (unsigned short)(u >> 16);
}

__global__ __launch_bounds__(256)
void node_edge_wmma_kernel(const float* __restrict__ x,
                           const float* __restrict__ inci,
                           const float* __restrict__ w,
                           const float* __restrict__ bias,
                           float* __restrict__ out)
{
    __shared__ unsigned short As[TN][KP];   // bf16(w*inci+b) tile, [row][k]
    __shared__ unsigned short Bs[TC][KP];   // bf16(x) tile TRANSPOSED, [col(f)][k]

    const int tid  = threadIdx.x;
    const int n0   = blockIdx.x * TN;            // node-row tile base
    const int c0   = blockIdx.y * TC;            // flattened (batch,feature) column base
    const int bat  = c0 / F_;                    // 128 | 256 -> tile is within one batch
    const int f0   = c0 % F_;

    const int wave   = tid >> 5;
    const int lane   = tid & 31;
    const int laneLo = lane & 15;
    const int laneHi = lane >> 4;                // 0 or 1: selects K half per ISA layouts
    const int wr     = wave >> 1;                // 0..3 : 32-row group
    const int wc     = wave & 1;                 // 0..1 : 64-col group

    v8f acc[2][4] = {};                          // 2x4 16x16 f32 accumulators

    const float* wP = w    + (long)n0 * E_;
    const float* iP = inci + (long)n0 * E_;
    const float* bP = bias + (long)n0 * E_;
    const float* xP = x + (long)bat * E_ * F_ + f0;

    for (int k0 = 0; k0 < E_; k0 += TK) {
        __syncthreads();

        // ---- Fill A tile: m = w*inci + b, cast to bf16. 128x64 elems, float4 per thread x8.
        #pragma unroll
        for (int it = 0; it < 8; ++it) {
            const int idx = it * 256 + tid;
            const int kq  = (idx & 15) * 4;      // 0..60
            const int row = idx >> 4;            // 0..127
            const long g  = (long)row * E_ + (k0 + kq);
            const float4 wv = *(const float4*)(wP + g);
            const float4 iv = *(const float4*)(iP + g);
            const float4 bv = *(const float4*)(bP + g);
            union { unsigned short h[4]; uint2 u; } pk;
            pk.h[0] = f32_to_bf16(fmaf(wv.x, iv.x, bv.x));
            pk.h[1] = f32_to_bf16(fmaf(wv.y, iv.y, bv.y));
            pk.h[2] = f32_to_bf16(fmaf(wv.z, iv.z, bv.z));
            pk.h[3] = f32_to_bf16(fmaf(wv.w, iv.w, bv.w));
            *(uint2*)&As[row][kq] = pk.u;        // ds_store_b64
        }

        // ---- Fill B tile transposed: x[bat, k0+krow, f0+fq..] -> Bs[f][k]. 64x128 elems.
        #pragma unroll
        for (int it = 0; it < 8; ++it) {
            const int idx  = it * 256 + tid;
            const int fq   = (idx & 31) * 4;     // 0..124
            const int krow = idx >> 5;           // 0..63
            const float4 xv = *(const float4*)(xP + (long)(k0 + krow) * F_ + fq);
            Bs[fq + 0][krow] = f32_to_bf16(xv.x);
            Bs[fq + 1][krow] = f32_to_bf16(xv.y);
            Bs[fq + 2][krow] = f32_to_bf16(xv.z);
            Bs[fq + 3][krow] = f32_to_bf16(xv.w);
        }
        __syncthreads();

        // ---- Speculative prefetch of next K-chunk (global_prefetch_b8)
        if (k0 + TK < E_) {
            const int kq  = (tid & 15) * 4;
            const int row = tid >> 4;
            const long g  = (long)row * E_ + (k0 + TK + kq);
            __builtin_prefetch(wP + g, 0, 1);
            __builtin_prefetch(iP + g, 0, 1);
            __builtin_prefetch(bP + g, 0, 1);
            __builtin_prefetch(xP + (long)(k0 + TK + (tid >> 5)) * F_ + (tid & 31) * 4, 0, 1);
        }

        // ---- Two WMMA K-steps of 32 over the 64-wide chunk
        #pragma unroll
        for (int ks = 0; ks < TK; ks += 32) {
            // B fragments: 32x16 bf16. lanes0-15: K=ks+0..7 then ks+16..23 (col = laneLo),
            // lanes16-31: K=ks+8..15 then ks+24..31 -> two 16B loads from transposed tile.
            Frag32 bf[4];
            #pragma unroll
            for (int j = 0; j < 4; ++j) {
                const int col = wc * 64 + j * 16 + laneLo;
                bf[4 - 4 + j].q[0] = *(const uint4*)&Bs[col][ks + laneHi * 8];
                bf[j].q[1] = *(const uint4*)&Bs[col][ks + 16 + laneHi * 8];
            }
            #pragma unroll
            for (int i = 0; i < 2; ++i) {
                // A fragment: 16x32 bf16. lanes0-15: K=ks+0..7 then ks+16..23,
                // lanes16-31: K=ks+8..15 then ks+24..31 (row = laneLo).
                const int row = wr * 32 + i * 16 + laneLo;
                Frag32 af;
                af.q[0] = *(const uint4*)&As[row][ks + laneHi * 8];
                af.q[1] = *(const uint4*)&As[row][ks + 16 + laneHi * 8];
                #pragma unroll
                for (int j = 0; j < 4; ++j) {
                    acc[i][j] = __builtin_amdgcn_wmma_f32_16x16x32_bf16(
                        /*neg_a=*/false, af.v, /*neg_b=*/false, bf[j].v,
                        /*c_mod=*/(short)0, acc[i][j],
                        /*reuse_a=*/false, /*reuse_b=*/false);
                }
            }
        }
    }

    // ---- Epilogue: C/D layout — VGPR r: lanes0-15 M=r, lanes16-31 M=r+8, N=laneLo.
    float* o = out + (long)bat * N_ * F_;
    #pragma unroll
    for (int i = 0; i < 2; ++i) {
        #pragma unroll
        for (int j = 0; j < 4; ++j) {
            const int col = f0 + wc * 64 + j * 16 + laneLo;
            #pragma unroll
            for (int r = 0; r < 8; ++r) {
                const int row = n0 + wr * 32 + i * 16 + r + laneHi * 8;
                o[(long)row * F_ + col] = acc[i][j][r];
            }
        }
    }
}

extern "C" void kernel_launch(void* const* d_in, const int* in_sizes, int n_in,
                              void* d_out, int out_size, void* d_ws, size_t ws_size,
                              hipStream_t stream) {
    const float* x    = (const float*)d_in[0];
    const float* inci = (const float*)d_in[1];
    const float* w    = (const float*)d_in[2];
    const float* bias = (const float*)d_in[3];
    float* out        = (float*)d_out;

    dim3 grid(N_ / TN, (B_ * F_) / TC);   // 32 x 32 workgroups
    dim3 block(256);                       // 8 wave32 waves
    node_edge_wmma_kernel<<<grid, block, 0, stream>>>(x, inci, w, bias, out);
}
